// PrefixAttnProbe_68015102099932
// MI455X (gfx1250) — matrix operand order repaired and checked by reading, not compiled, not run on hardware
//
#include <hip/hip_runtime.h>
#include <hip/hip_bf16.h>

// ---------------------------------------------------------------------------
// Problem constants (from reference): B=2, T=2048, HID=1024, KV=934, V=32000
// KV padded to 960 so all GEMM dims are multiples of 32/64. Padding is zero-
// filled so it contributes nothing to any dot product.
// ---------------------------------------------------------------------------
#define BN    2
#define TT    2048
#define HID   1024
#define KV    934
#define KVP   960
#define VOC   32000
#define MROWS (BN * TT)   // 4096

typedef __bf16 bf16_t;
typedef __attribute__((ext_vector_type(16))) __bf16 v16bf;
typedef __attribute__((ext_vector_type(8)))  float  v8f;

static __device__ __forceinline__ v8f wmma_bf16(v16bf a, v16bf b, v8f c) {
    // (neg_a, A, neg_b, B, c_mod, C, reuse_a, reuse_b)
    return __builtin_amdgcn_wmma_f32_16x16x32_bf16(false, a, false, b,
                                                   (short)0, c, false, false);
}

// A-fragment (16x32 bf16, row-major source):
// lane = m + 16*h ; vector elems 0..7 = K[8h..8h+7], elems 8..15 = K[16+8h..16+8h+7]
static __device__ __forceinline__ v16bf load_a_frag(const bf16_t* row, int k0, int lh) {
    v16bf v;
    uint4* p = reinterpret_cast<uint4*>(&v);
    p[0] = *reinterpret_cast<const uint4*>(row + k0 + 8 * lh);
    p[1] = *reinterpret_cast<const uint4*>(row + k0 + 16 + 8 * lh);
    return v;
}

// B-fragment (32x16 bf16) from transposed (row-major [N,K]) source:
// lane = n + 16*h ; vector elems 0..7 = K[8h..8h+7], elems 8..15 = K[16+8h..16+8h+7]
static __device__ __forceinline__ v16bf load_b_frag(const bf16_t* row, int k0, int lh) {
    v16bf v;
    uint4* p = reinterpret_cast<uint4*>(&v);
    const bf16_t* s = row + k0 + 8 * lh;
    p[0] = reinterpret_cast<const uint4*>(s)[0];
    p[1] = reinterpret_cast<const uint4*>(s + 16)[0];
    return v;
}

// ---------------------------------------------------------------------------
// K1a: fp32 -> bf16 elementwise
// ---------------------------------------------------------------------------
__global__ __launch_bounds__(256) void f32_to_bf16_kernel(const float* __restrict__ in,
                                                          bf16_t* __restrict__ out, int n) {
    for (int i = blockIdx.x * blockDim.x + threadIdx.x; i < n; i += gridDim.x * blockDim.x)
        out[i] = (bf16_t)in[i];
}

// ---------------------------------------------------------------------------
// K1b: transpose + zero-pad + bf16-convert:  out[n][k] = in[k][n] (0 if OOB)
// in: [K_in x N_in] fp32, out: [N_out x K_out] bf16. LDS-tiled 32x32.
// ---------------------------------------------------------------------------
__global__ __launch_bounds__(256) void transpose_pad_kernel(const float* __restrict__ in,
                                                            bf16_t* __restrict__ out,
                                                            int K_in, int N_in,
                                                            int K_out, int N_out) {
    __shared__ float tile[32][33];
    int k0 = blockIdx.x * 32, n0 = blockIdx.y * 32;
    int tx = threadIdx.x, ty = threadIdx.y;   // block (32, 8)
    #pragma unroll
    for (int j = 0; j < 4; ++j) {
        int k = k0 + ty + 8 * j, n = n0 + tx;
        float v = 0.0f;
        if (k < K_in && n < N_in) v = in[(size_t)k * N_in + n];
        tile[ty + 8 * j][tx] = v;
    }
    __syncthreads();
    #pragma unroll
    for (int j = 0; j < 4; ++j) {
        int n = n0 + ty + 8 * j, k = k0 + tx;
        if (n < N_out && k < K_out)
            out[(size_t)n * K_out + k] = (bf16_t)tile[tx][ty + 8 * j];
    }
}

// ---------------------------------------------------------------------------
// K2: bf16 WMMA GEMM  C[M x N] = A[M x Kd] * Bt[N x Kd]^T  (+bias)
// Block: 256 threads = 8 waves arranged 4(M) x 2(N); wave tile 64x32
// (acc[4][2] v8f), block tile 256 x 64.  Per k-step: 6 fragment loads feed
// 8 WMMAs (0.75 loads/WMMA) -- the kernel is matrix-issue bound since both
// operands of the dominant vocab GEMM live in the 192MB L2.
// Launcher guarantees: M % 256 == 0, N % 64 == 0, Kd % 32 == 0.
// ---------------------------------------------------------------------------
template <bool OUT_BF16, bool TRANS_OUT, bool BIAS>
__global__ __launch_bounds__(256) void gemm_bf16_kernel(
    const bf16_t* __restrict__ A, long long sA,
    const bf16_t* __restrict__ Bt, long long sB,
    void* __restrict__ Cv, long long sC,
    const float* __restrict__ bias,
    int Kd, int lda, int ldb, int ldc) {

    A  += (size_t)blockIdx.z * sA;
    Bt += (size_t)blockIdx.z * sB;
    bf16_t* Cb = (bf16_t*)Cv + (size_t)blockIdx.z * sC;
    float*  Cf = (float*)Cv  + (size_t)blockIdx.z * sC;

    const int tid  = threadIdx.x;
    const int wave = tid >> 5;
    const int lane = tid & 31;
    const int lm   = lane & 15;
    const int lh   = lane >> 4;

    const int m0 = blockIdx.y * 256 + (wave >> 1) * 64;
    const int n0 = blockIdx.x * 64  + (wave & 1) * 32;

    const bf16_t* ap[4];
    #pragma unroll
    for (int i = 0; i < 4; ++i)
        ap[i] = A + (size_t)(m0 + i * 16 + lm) * lda;
    const bf16_t* b0p = Bt + (size_t)(n0 + lm) * ldb;
    const bf16_t* b1p = Bt + (size_t)(n0 + 16 + lm) * ldb;

    v8f acc[4][2] = {};

    for (int k0 = 0; k0 < Kd; k0 += 32) {
        v16bf bv0 = load_b_frag(b0p, k0, lh);
        v16bf bv1 = load_b_frag(b1p, k0, lh);
        #pragma unroll
        for (int i = 0; i < 4; ++i) {
            v16bf av = load_a_frag(ap[i], k0, lh);
            acc[i][0] = wmma_bf16(av, bv0, acc[i][0]);
            acc[i][1] = wmma_bf16(av, bv1, acc[i][1]);
        }
    }

    float bia[2] = {0.0f, 0.0f};
    if (BIAS) { bia[0] = bias[n0 + lm]; bia[1] = bias[n0 + 16 + lm]; }

    #pragma unroll
    for (int i = 0; i < 4; ++i)
        #pragma unroll
        for (int j = 0; j < 2; ++j)
            #pragma unroll
            for (int r = 0; r < 8; ++r) {
                int m = m0 + i * 16 + r + 8 * lh;
                int n = n0 + j * 16 + lm;
                float v = acc[i][j][r] + (BIAS ? bia[j] : 0.0f);
                size_t idx = TRANS_OUT ? ((size_t)n * ldc + m) : ((size_t)m * ldc + n);
                if (OUT_BF16) Cb[idx] = (bf16_t)v;
                else          Cf[idx] = v;
            }
}

// ---------------------------------------------------------------------------
// K3: scores + masked softmax.  One block per (batch, 32-query tile).
// LDS: S[32][2048] fp32 (256KB) + staged Q tile [32][960] bf16 (60KB)
//   -> 316KB, possible only with CDNA5's 320KB-per-WGP LDS.
// S tile computed with WMMA (Q from LDS, K rows from global = Bt layout),
// then causal+pad mask, wave shfl reductions, exp, P written bf16.
// ---------------------------------------------------------------------------
__global__ __launch_bounds__(256) void scores_softmax_kernel(
    const bf16_t* __restrict__ qg, const bf16_t* __restrict__ kg,
    const int* __restrict__ mask, bf16_t* __restrict__ P) {

    extern __shared__ char smem[];
    float*  S  = (float*)smem;                                   // 32 x 2048
    bf16_t* Qs = (bf16_t*)(smem + 32 * TT * sizeof(float));      // 32 x 960

    const int b  = blockIdx.y;
    const int q0 = blockIdx.x * 32;
    const int tid = threadIdx.x;

    // Stage Q tile (contiguous 32*960 bf16 = 3840 x uint4)
    {
        const uint4* src = (const uint4*)(qg + (size_t)(b * TT + q0) * KVP);
        uint4* dst = (uint4*)Qs;
        for (int i = tid; i < 32 * KVP / 8; i += 256) dst[i] = src[i];
    }
    __syncthreads();

    const int wave = tid >> 5;
    const int lane = tid & 31;
    const int lm = lane & 15;
    const int lh = lane >> 4;
    const int mi = wave >> 2;   // 0..1 -> query sub-tile of 16
    const int ni = wave & 3;    // 0..3 -> key sub-tile within 64-chunk
    const float SCALE = 0.03272155946f;   // 1/sqrt(934)

    const bf16_t* arow = Qs + (size_t)(mi * 16 + lm) * KVP;
    for (int c = 0; c < TT / 64; ++c) {
        const int kb = c * 64 + ni * 16;
        const bf16_t* brow = kg + (size_t)(b * TT + kb + lm) * KVP;
        v8f acc = {};
        for (int k0 = 0; k0 < KVP; k0 += 32) {
            v16bf av = load_a_frag(arow, k0, lh);   // ds_load_b128 x2
            v16bf bv = load_b_frag(brow, k0, lh);   // global_load_b128 x2
            acc = wmma_bf16(av, bv, acc);
        }
        #pragma unroll
        for (int r = 0; r < 8; ++r)
            S[(mi * 16 + r + 8 * lh) * TT + kb + lm] = acc[r] * SCALE;
    }
    __syncthreads();

    // Softmax: each wave owns 4 rows
    for (int rr = 0; rr < 4; ++rr) {
        const int row = wave * 4 + rr;
        const int qgl = q0 + row;                 // global query index
        const float NEG = -3.0e38f;

        float mx = NEG;
        for (int s = lane; s < TT; s += 32) {
            float x = S[row * TT + s];
            bool valid = (s <= qgl) && (mask[b * TT + s] != 0);
            x = valid ? x : NEG;
            S[row * TT + s] = x;
            mx = fmaxf(mx, x);
        }
        #pragma unroll
        for (int off = 16; off > 0; off >>= 1)
            mx = fmaxf(mx, __shfl_xor(mx, off, 32));

        float sum = 0.0f;
        for (int s = lane; s < TT; s += 32) {
            float e = __expf(S[row * TT + s] - mx);
            S[row * TT + s] = e;
            sum += e;
        }
        #pragma unroll
        for (int off = 16; off > 0; off >>= 1)
            sum += __shfl_xor(sum, off, 32);

        // all-masked row -> zeros (reference nan_to_num)
        float inv = (mx > -1.0e38f) ? (1.0f / sum) : 0.0f;
        bf16_t* prow = P + (size_t)(b * TT + qgl) * TT;
        for (int s = lane; s < TT; s += 32)
            prow[s] = (bf16_t)(S[row * TT + s] * inv);
    }
}

// ---------------------------------------------------------------------------
// Host launcher
// ---------------------------------------------------------------------------
extern "C" void kernel_launch(void* const* d_in, const int* in_sizes, int n_in,
                              void* d_out, int out_size, void* d_ws, size_t ws_size,
                              hipStream_t stream) {
    (void)in_sizes; (void)n_in; (void)out_size; (void)ws_size;

    const float* h    = (const float*)d_in[0];
    const int*   mask = (const int*)d_in[1];
    const float* Wq   = (const float*)d_in[2];
    const float* Wk   = (const float*)d_in[3];
    const float* Wv   = (const float*)d_in[4];
    const float* Wo   = (const float*)d_in[5];
    const float* bo   = (const float*)d_in[6];
    float* out = (float*)d_out;

    // Workspace carve-up (~124 MB total)
    char* ws = (char*)d_ws;
    size_t off = 0;
    auto take = [&](size_t bytes) {
        char* p = ws + off;
        off += (bytes + 255) & ~(size_t)255;
        return p;
    };
    bf16_t* h_bf  = (bf16_t*)take((size_t)MROWS * HID * 2);
    bf16_t* Wq_t  = (bf16_t*)take((size_t)KVP * HID * 2);
    bf16_t* Wk_t  = (bf16_t*)take((size_t)KVP * HID * 2);
    bf16_t* Wv_t  = (bf16_t*)take((size_t)KVP * HID * 2);
    bf16_t* Wo_t  = (bf16_t*)take((size_t)VOC * KVP * 2);
    bf16_t* q_bf  = (bf16_t*)take((size_t)MROWS * KVP * 2);
    bf16_t* k_bf  = (bf16_t*)take((size_t)MROWS * KVP * 2);
    bf16_t* vT_bf = (bf16_t*)take((size_t)KVP * MROWS * 2);   // [960][4096]
    bf16_t* Pbuf  = (bf16_t*)take((size_t)BN * TT * TT * 2);  // [B][T][T]
    bf16_t* ctx_bf= (bf16_t*)take((size_t)MROWS * KVP * 2);

    // --- 1. conversions / transposes --------------------------------------
    f32_to_bf16_kernel<<<4096, 256, 0, stream>>>(h, h_bf, MROWS * HID);

    transpose_pad_kernel<<<dim3(HID / 32, KVP / 32), dim3(32, 8), 0, stream>>>(
        Wq, Wq_t, HID, KV, HID, KVP);
    transpose_pad_kernel<<<dim3(HID / 32, KVP / 32), dim3(32, 8), 0, stream>>>(
        Wk, Wk_t, HID, KV, HID, KVP);
    transpose_pad_kernel<<<dim3(HID / 32, KVP / 32), dim3(32, 8), 0, stream>>>(
        Wv, Wv_t, HID, KV, HID, KVP);
    transpose_pad_kernel<<<dim3(KVP / 32, VOC / 32), dim3(32, 8), 0, stream>>>(
        Wo, Wo_t, KV, VOC, KVP, VOC);

    // --- 2. QKV projections (bf16 WMMA GEMMs) -----------------------------
    dim3 gqkv(KVP / 64, MROWS / 256, 1);
    gemm_bf16_kernel<true, false, false><<<gqkv, 256, 0, stream>>>(
        h_bf, 0, Wq_t, 0, q_bf, 0, nullptr, HID, HID, HID, KVP);
    gemm_bf16_kernel<true, false, false><<<gqkv, 256, 0, stream>>>(
        h_bf, 0, Wk_t, 0, k_bf, 0, nullptr, HID, HID, HID, KVP);
    gemm_bf16_kernel<true, true, false><<<gqkv, 256, 0, stream>>>(
        h_bf, 0, Wv_t, 0, vT_bf, 0, nullptr, HID, HID, HID, MROWS);

    // --- 3. scores + masked softmax ---------------------------------------
    const int SMEM = 32 * TT * (int)sizeof(float) + 32 * KVP * 2;  // 323584 B
    hipFuncSetAttribute((const void*)scores_softmax_kernel,
                        hipFuncAttributeMaxDynamicSharedMemorySize, SMEM);
    scores_softmax_kernel<<<dim3(TT / 32, BN), 256, SMEM, stream>>>(
        q_bf, k_bf, mask, Pbuf);

    // --- 4. ctx = P @ V (batched) -----------------------------------------
    gemm_bf16_kernel<true, false, false><<<dim3(KVP / 64, TT / 256, BN), 256, 0, stream>>>(
        Pbuf, (long long)TT * TT, vT_bf, (long long)TT,
        ctx_bf, (long long)TT * KVP, nullptr, TT, TT, MROWS, KVP);

    // --- 5. out = ctx @ Wo + bo (fp32 output, dominant GEMM) --------------
    gemm_bf16_kernel<false, false, true><<<dim3(VOC / 64, MROWS / 256), 256, 0, stream>>>(
        ctx_bf, 0, Wo_t, 0, out, 0, bo, KVP, KVP, KVP, VOC);
}
